// NoisyTopkRouter_86878598464359
// MI455X (gfx1250) — matrix operand order repaired and checked by reading, not compiled, not run on hardware
//
#include <hip/hip_runtime.h>
#include <hip/hip_bf16.h>

// ---------------------------------------------------------------------------
// NoisyTopkRouter fused kernel for gfx1250 (MI455X)
//   logits  = h @ Ww^T + bw            [32768 x 64]
//   npre    = h @ Wn^T + bn            [32768 x 64]
//   noisy   = logits + noise * softplus(npre)
//   full    = softmax(noisy)
//   top8    -> ix, sparse softmax -> gates
// Both GEMMs fused: B matrix = concat(Ww, Wn) -> 128 output columns.
// Streams h (f32, 512 MB) once, converts to bf16, v_wmma_f32_16x16x32_bf16.
// A sched_barrier(0) between the fragment loads and the WMMA burst keeps the
// 18 ds_load_b128 of each k-step as one clause (single s_wait_dscnt) and
// forces the 8 B fragments into distinct registers so the 8 WMMAs issue
// back-to-back.
// ---------------------------------------------------------------------------

#if defined(__has_builtin)
#  if __has_builtin(__builtin_amdgcn_sched_barrier)
#    define SCHED_FENCE() __builtin_amdgcn_sched_barrier(0)
#  endif
#endif
#ifndef SCHED_FENCE
#  define SCHED_FENCE()
#endif

typedef __attribute__((ext_vector_type(16))) __bf16 v16bf;
typedef __attribute__((ext_vector_type(8)))  float  v8f;

namespace {
constexpr int Dd = 4096;          // hidden dim
constexpr int Ee = 64;            // experts
constexpr int NROWS = 8 * 4096;   // B*S = 32768
constexpr int MT = 128;           // rows per block
constexpr int KC = 64;            // K chunk (2 wmma k-steps)
constexpr int AST = 72;           // LDS row stride in bf16 (72*2=144B, 16B aligned, padded)
constexpr int CST = 132;          // LDS epilogue row stride in f32 (padded)
constexpr size_t GATES_OFF = 0;
constexpr size_t IX_OFF    = (size_t)NROWS * Ee;            // 2,097,152 floats
constexpr size_t FULL_OFF  = IX_OFF + (size_t)NROWS * 8;    // 2,359,296 floats
}

union Frag { v16bf v; uint4 u[2]; };

__device__ __forceinline__ unsigned short f32_to_bf16_rne(float f) {
  unsigned int u = __float_as_uint(f);
  unsigned int r = u + 0x7FFFu + ((u >> 16) & 1u);
  return (unsigned short)(r >> 16);
}

__global__ __launch_bounds__(256, 2)
void noisy_topk_router_kernel(const float* __restrict__ h,
                              const float* __restrict__ Ww,
                              const float* __restrict__ bw,
                              const float* __restrict__ Wn,
                              const float* __restrict__ bn,
                              const float* __restrict__ noise,
                              float* __restrict__ out) {
  __shared__ __align__(16) char smem[MT * CST * 4];   // 67,584 B; aliased
  unsigned short* As = (unsigned short*)smem;                  // [128][72] bf16
  unsigned short* Bs = (unsigned short*)(smem + MT * AST * 2); // [128][72] bf16
  float* Cs = (float*)smem;                                    // [128][132] f32 (after barrier)

  const int t    = threadIdx.x;
  const int wv   = t >> 5;
  const int lane = t & 31;
  const int row0 = blockIdx.x * MT;

  v8f acc[8] = {};   // 8 column tiles (128 cols) x 16x16 f32 -> 64 VGPRs

  for (int k0 = 0; k0 < Dd; k0 += KC) {
    __syncthreads();
    // ---- stage A: h[row0..row0+127][k0..k0+63] f32 -> bf16 LDS (coalesced) --
    #pragma unroll
    for (int i = 0; i < 8; ++i) {
      int lin = i * 256 + t;                 // 0..2047 float4 slots
      int r = lin >> 4, g = lin & 15;        // 16 float4 per row
      float4 va = *(const float4*)(h + (size_t)(row0 + r) * Dd + k0 + g * 4);
      unsigned short* d = As + r * AST + g * 4;
      d[0] = f32_to_bf16_rne(va.x); d[1] = f32_to_bf16_rne(va.y);
      d[2] = f32_to_bf16_rne(va.z); d[3] = f32_to_bf16_rne(va.w);
    }
    // ---- stage B: cols 0..63 = Ww experts, cols 64..127 = Wn experts -------
    #pragma unroll
    for (int i = 0; i < 8; ++i) {
      int lin = i * 256 + t;
      int c = lin >> 4, g = lin & 15;
      const float* src = (c < Ee) ? (Ww + (size_t)c * Dd) : (Wn + (size_t)(c - Ee) * Dd);
      float4 vb = *(const float4*)(src + k0 + g * 4);
      unsigned short* d = Bs + c * AST + g * 4;
      d[0] = f32_to_bf16_rne(vb.x); d[1] = f32_to_bf16_rne(vb.y);
      d[2] = f32_to_bf16_rne(vb.z); d[3] = f32_to_bf16_rne(vb.w);
    }
    // ---- prefetch next h chunk (global_prefetch_b8) ------------------------
    if (k0 + KC < Dd) {
      const float* pf = h + (size_t)(row0 + (t >> 1)) * Dd + (k0 + KC) + (t & 1) * 32;
      __builtin_prefetch(pf, 0, 1);
    }
    __syncthreads();

    // ---- 2 k-steps; per step: 18 ds_load_b128 clause, then 8 WMMAs ---------
    const int m  = lane & 15;
    const int hi = lane >> 4;        // lane half selects K sub-block
    #pragma unroll
    for (int s = 0; s < 2; ++s) {
      // A 16x32 bf16 layout: lanes<16: K {0..7,16..23}; lanes>=16: K {8..15,24..31}
      Frag a;
      const unsigned short* ap = As + (wv * 16 + m) * AST + s * 32 + hi * 8;
      a.u[0] = *(const uint4*)(ap);
      a.u[1] = *(const uint4*)(ap + 16);
      // B 32x16 bf16 layout: lanes<16: K {0..7,16..23}; lanes>=16: K {8..15,24..31} (N = lane%16)
      Frag b[8];
      #pragma unroll
      for (int tc = 0; tc < 8; ++tc) {
        const unsigned short* bp = Bs + (tc * 16 + m) * AST + s * 32 + hi * 8;
        b[tc].u[0] = *(const uint4*)(bp);
        b[tc].u[1] = *(const uint4*)(bp + 16);
      }
      // Nothing may cross this point: keeps all 18 LDS loads above (single
      // dscnt wait), all 8 WMMAs below, and forces the 8 B fragments to be
      // simultaneously live -> distinct registers, back-to-back WMMA issue.
      // Next k-step's loads start after this step's WMMAs in program order,
      // so they may still hoist into the WMMA burst (they never cross a
      // barrier), overlapping LDS latency with the matrix pipe.
      SCHED_FENCE();
      #pragma unroll
      for (int tc = 0; tc < 8; ++tc) {
        acc[tc] = __builtin_amdgcn_wmma_f32_16x16x32_bf16(
            false, a.v, false, b[tc].v, (short)0, acc[tc], false, false);
      }
    }
  }

  // ---- spill accumulators to LDS per C-matrix layout ------------------------
  __syncthreads();   // staging space is dead now; reuse as Cs
  {
    const int n  = lane & 15;
    const int rb = wv * 16 + ((lane >> 4) << 3);  // VGPR j -> row rb+j
    #pragma unroll
    for (int tc = 0; tc < 8; ++tc)
      #pragma unroll
      for (int j = 0; j < 8; ++j)
        Cs[(rb + j) * CST + tc * 16 + n] = acc[tc][j];
  }
  __syncthreads();

  // ---- per-row epilogue: one thread per row --------------------------------
  if (t < MT) {
    const int gr = row0 + t;
    float* crow = Cs + t * CST;
    const float* nrow = noise + (size_t)gr * Ee;

    // pass 1: noisy logits + running max (store noisy in crow[0..63])
    float mx = -INFINITY;
    for (int e = 0; e < Ee; ++e) {
      float lg = crow[e] + bw[e];
      float np = crow[Ee + e] + bn[e];
      float sc = (np > 20.f) ? np : log1pf(__expf(np));   // softplus
      float ny = lg + nrow[e] * sc;
      crow[e] = ny;
      mx = fmaxf(mx, ny);
    }

    // pass 2: exp/sum for full softmax + top-8 insertion select (stable ties)
    float tv[8]; int ti[8];
    #pragma unroll
    for (int i = 0; i < 8; ++i) { tv[i] = -INFINITY; ti[i] = 0; }
    float sum = 0.f;
    for (int e = 0; e < Ee; ++e) {
      float ny = crow[e];
      float ex = __expf(ny - mx);
      crow[Ee + e] = ex;
      sum += ex;
      if (ny > tv[7]) {
        int p = 7;
        while (p > 0 && ny > tv[p - 1]) { tv[p] = tv[p - 1]; ti[p] = ti[p - 1]; --p; }
        tv[p] = ny; ti[p] = e;
      }
    }
    const float inv = 1.f / sum;

    float* gatep = out + GATES_OFF + (size_t)gr * Ee;
    float* fullp = out + FULL_OFF  + (size_t)gr * Ee;
    for (int e = 0; e < Ee; ++e) {
      fullp[e] = crow[Ee + e] * inv;
      gatep[e] = 0.f;                       // softmax(-inf) == 0 off the top-k
    }

    // sparse softmax over the 8 kept logits
    const float m2 = tv[0];
    float s2 = 0.f;
    #pragma unroll
    for (int i = 0; i < 8; ++i) s2 += __expf(tv[i] - m2);
    const float inv2 = 1.f / s2;
    int* ixp = (int*)(out + IX_OFF) + (size_t)gr * 8;
    #pragma unroll
    for (int i = 0; i < 8; ++i) {
      gatep[ti[i]] = __expf(tv[i] - m2) * inv2;
      ixp[i] = ti[i];
    }
  }
}

extern "C" void kernel_launch(void* const* d_in, const int* in_sizes, int n_in,
                              void* d_out, int out_size, void* d_ws, size_t ws_size,
                              hipStream_t stream) {
  (void)in_sizes; (void)n_in; (void)d_ws; (void)ws_size; (void)out_size;
  const float* h     = (const float*)d_in[0];
  const float* Ww    = (const float*)d_in[1];
  const float* bw    = (const float*)d_in[2];
  const float* Wn    = (const float*)d_in[3];
  const float* bn    = (const float*)d_in[4];
  const float* noise = (const float*)d_in[5];
  float* out = (float*)d_out;

  dim3 grid(NROWS / MT);   // 256 blocks
  dim3 block(256);         // 8 wave32s
  noisy_topk_router_kernel<<<grid, block, 0, stream>>>(h, Ww, bw, Wn, bn, noise, out);
}
